// QWindowAttention_34376918237364
// MI455X (gfx1250) — 16-bit and 8-bit matrix operand order repaired and checked by reading, compiled, not run on hardware
//
#include <hip/hip_runtime.h>

typedef __attribute__((ext_vector_type(16))) _Float16 v16h;
typedef __attribute__((ext_vector_type(8)))  float    v8f;
typedef __attribute__((ext_vector_type(8)))  int      v8i;

#define NTOK  49
#define NP    64
#define HEADS 8
#define DIM   256
#define HD    32

__device__ __forceinline__ int qsym_r(float v, float rs) {
    float r = rintf(v * rs);
    r = fminf(fmaxf(r, -5.f), 4.f);
    return (int)r;
}
__device__ __forceinline__ int qasym_r(float v, float rs) {
    float r = rintf(v * rs);
    r = fminf(fmaxf(r, 0.f), 9.f);
    return (int)r;
}

__global__ void qwa_prep_weights(const float* __restrict__ qkv_w,
                                 const float* __restrict__ proj_w,
                                 _Float16* __restrict__ wq,
                                 _Float16* __restrict__ wp) {
    int idx = blockIdx.x * blockDim.x + threadIdx.x;
    if (idx < 768 * 256) wq[idx] = (_Float16)qkv_w[idx];
    if (idx < 256 * 256) wp[idx] = (_Float16)proj_w[idx];
}

__global__ __launch_bounds__(256)
void qwa_fused(const float* __restrict__ x,
               const float* __restrict__ mask,
               const _Float16* __restrict__ wq,
               const float* __restrict__ qkv_b,
               const _Float16* __restrict__ wp,
               const float* __restrict__ proj_b,
               const float* __restrict__ rel_tab,
               const float* __restrict__ ps_q, const float* __restrict__ ps_k,
               const float* __restrict__ ps_v, const float* __restrict__ ps_attn,
               const float* __restrict__ ps_after,
               float* __restrict__ out) {
    __shared__ _Float16     xh[NP * DIM];
    __shared__ signed char  q8[HEADS * NP * HD];
    __shared__ signed char  k8[HEADS * NP * HD];
    __shared__ signed char  v8t[HEADS * HD * NP];
    __shared__ unsigned char a8[HEADS * NP * NP];

    const int tid   = threadIdx.x;
    const int lane  = tid & 31;
    const int wv    = __builtin_amdgcn_readfirstlane(tid >> 5);
    const int ln    = lane & 15;
    const int lhalf = (lane < 16) ? 0 : 1;
    const int b     = blockIdx.x;

    const float sq = ps_q[0], sk = ps_k[0], sv = ps_v[0];
    const float sa = ps_attn[0], sf = ps_after[0];
    const float rsq = __builtin_amdgcn_rcpf(sq);
    const float rsk = __builtin_amdgcn_rcpf(sk);
    const float rsv = __builtin_amdgcn_rcpf(sv);
    const float rsa49 = __builtin_amdgcn_rcpf(49.0f * sa);
    const float rsf = __builtin_amdgcn_rcpf(sf);
    const float scale = 0.17677669529663687f;

    const float4* xin = (const float4*)(x + (size_t)b * NTOK * DIM);
    for (int i = tid; i < NP * DIM / 4; i += 256) {
        float4 f = {0.f, 0.f, 0.f, 0.f};
        if (i < NTOK * DIM / 4) f = xin[i];
        _Float16* d = &xh[i * 4];
        d[0] = (_Float16)f.x; d[1] = (_Float16)f.y;
        d[2] = (_Float16)f.z; d[3] = (_Float16)f.w;
    }
    __syncthreads();

    for (int t = wv; t < 4 * 48; t += 8) {
        const int tm = t & 3;
        const int tn = t >> 2;
        v8f acc = {0.f,0.f,0.f,0.f,0.f,0.f,0.f,0.f};
        const _Float16* abase = &xh[(tm * 16 + ln) * DIM];
        const _Float16* bbase = &wq[(size_t)(tn * 16 + ln) * DIM];
        for (int kk = 0; kk < 8; ++kk) {
            const int k0 = kk * 32;
            v16h af, bf;
            af = *(const v16h*)(abase + k0 + 16 * lhalf);
            bf = *(const v16h*)(bbase + k0 + 16 * lhalf);
            acc = __builtin_amdgcn_wmma_f32_16x16x32_f16(false, af, false, bf,
                                                         (short)0, acc, false, false);
        }
        const int which = tn >> 4;
        const int h     = (tn >> 1) & 7;
        const int d     = ((tn & 1) << 4) + ln;
        const float bias = qkv_b[tn * 16 + ln];
        const float rs   = (which == 0) ? rsq : ((which == 1) ? rsk : rsv);
        if (which == 0) {
            for (int r = 0; r < 8; ++r) {
                const int row = tm * 16 + r + 8 * lhalf;
                q8[(h * NP + row) * HD + d] = (signed char)qsym_r(acc[r] + bias, rs);
            }
        } else if (which == 1) {
            for (int r = 0; r < 8; ++r) {
                const int row = tm * 16 + r + 8 * lhalf;
                k8[(h * NP + row) * HD + d] = (signed char)qsym_r(acc[r] + bias, rs);
            }
        } else {
            for (int r = 0; r < 8; ++r) {
                const int row = tm * 16 + r + 8 * lhalf;
                v8t[(h * HD + d) * NP + row] = (signed char)qsym_r(acc[r] + bias, rs);
            }
        }
    }
    __syncthreads();

    {
        const int h = wv;
        const int wi = b & 63;
        const float qkscale = sq * sk * scale;
        for (int tm = 0; tm < 4; ++tm) {
            v8i af = {0,0,0,0,0,0,0,0};
            const signed char* ab = &q8[(h * NP + tm * 16 + ln) * HD];
            const int ao = 8 * lhalf;
            af[0] = *(const int*)(ab + ao);      af[1] = *(const int*)(ab + ao + 4);
            af[2] = *(const int*)(ab + ao + 16); af[3] = *(const int*)(ab + ao + 20);
            for (int tn = 0; tn < 4; ++tn) {
                v8i bfr = {0,0,0,0,0,0,0,0};
                const signed char* bb = &k8[(h * NP + tn * 16 + ln) * HD];
                *((uint2*)&bfr) = *(const uint2*)(bb + 8 * lhalf); *(((uint2*)&bfr) + 1) = *(const uint2*)(bb + 16 + 8 * lhalf);
                v8i c0 = {0,0,0,0,0,0,0,0};
                v8i dd = __builtin_amdgcn_wmma_i32_16x16x64_iu8(true, af, true, bfr,
                                                                c0, false, false);
                const int j = tn * 16 + ln;
                const int jj = (j < NTOK) ? j : 0;
                const int rj = jj / 7, cj = jj - rj * 7;
                for (int r = 0; r < 8; ++r) {
                    const int i = tm * 16 + r + 8 * lhalf;
                    const int ii = (i < NTOK) ? i : 0;
                    const int ri = ii / 7, ci = ii - ri * 7;
                    const int relidx = (ri - rj + 6) * 13 + (ci - cj + 6);
                    float attn = (float)dd[r] * qkscale
                               + rel_tab[relidx * HEADS + h]
                               + mask[((size_t)wi * NTOK + ii) * NTOK + jj];
                    int u = qasym_r(attn, rsa49);
                    u = (i < NTOK && j < NTOK) ? u : 0;
                    a8[(h * NP + i) * NP + j] = (unsigned char)u;
                }
            }
        }
        __syncthreads();

        const float avscale = sa * sv;
        for (int tm = 0; tm < 4; ++tm) {
            v8i af;
            const unsigned char* ab = &a8[(h * NP + tm * 16 + ln) * NP];
            const int ao = 8 * lhalf;
            af[0] = *(const int*)(ab + ao);      af[1] = *(const int*)(ab + ao + 4);
            af[2] = *(const int*)(ab + ao + 16); af[3] = *(const int*)(ab + ao + 20);
            af[4] = *(const int*)(ab + ao + 32); af[5] = *(const int*)(ab + ao + 36);
            af[6] = *(const int*)(ab + ao + 48); af[7] = *(const int*)(ab + ao + 52);
            for (int tn = 0; tn < 2; ++tn) {
                v8i bfr;
                const int d = tn * 16 + ln;
                const signed char* bb = &v8t[(h * HD + d) * NP];
                *((uint2*)&bfr)       = *(const uint2*)(bb + 8 * lhalf);
                *(((uint2*)&bfr) + 1) = *(const uint2*)(bb + 16 + 8 * lhalf);
                *(((uint2*)&bfr) + 2) = *(const uint2*)(bb + 32 + 8 * lhalf);
                *(((uint2*)&bfr) + 3) = *(const uint2*)(bb + 48 + 8 * lhalf);
                v8i c0 = {0,0,0,0,0,0,0,0};
                v8i dd = __builtin_amdgcn_wmma_i32_16x16x64_iu8(false, af, true, bfr,
                                                                c0, false, false);
                for (int r = 0; r < 8; ++r) {
                    const int i = tm * 16 + r + 8 * lhalf;
                    const int iq = qsym_r((float)dd[r] * avscale, rsf);
                    xh[i * DIM + h * HD + d] = (_Float16)((float)iq * sf);
                }
            }
        }
    }
    __syncthreads();

    float* outb = out + (size_t)b * NTOK * DIM;
    for (int t = wv; t < 4 * 16; t += 8) {
        const int tm = t & 3;
        const int tn = t >> 2;
        v8f acc = {0.f,0.f,0.f,0.f,0.f,0.f,0.f,0.f};
        const _Float16* abase = &xh[(tm * 16 + ln) * DIM];
        const _Float16* bbase = &wp[(size_t)(tn * 16 + ln) * DIM];
        for (int kk = 0; kk < 8; ++kk) {
            const int k0 = kk * 32;
            v16h af, bf;
            af = *(const v16h*)(abase + k0 + 16 * lhalf);
            bf = *(const v16h*)(bbase + k0 + 16 * lhalf);
            acc = __builtin_amdgcn_wmma_f32_16x16x32_f16(false, af, false, bf,
                                                         (short)0, acc, false, false);
        }
        const int c = tn * 16 + ln;
        const float pb = proj_b[c];
        for (int r = 0; r < 8; ++r) {
            const int i = tm * 16 + r + 8 * lhalf;
            if (i < NTOK) outb[i * DIM + c] = acc[r] + pb;
        }
    }
}

extern "C" void kernel_launch(void* const* d_in, const int* in_sizes, int n_in,
                              void* d_out, int out_size, void* d_ws, size_t ws_size,
                              hipStream_t stream) {
    const float* x       = (const float*)d_in[0];
    const float* mask    = (const float*)d_in[1];
    const float* qkv_w   = (const float*)d_in[2];
    const float* qkv_b   = (const float*)d_in[3];
    const float* proj_w  = (const float*)d_in[4];
    const float* proj_b  = (const float*)d_in[5];
    const float* rel_tab = (const float*)d_in[6];
    const float* s_q     = (const float*)d_in[7];
    const float* s_k     = (const float*)d_in[8];
    const float* s_v     = (const float*)d_in[9];
    const float* s_attn  = (const float*)d_in[10];
    const float* s_after = (const float*)d_in[11];

    _Float16* wq = (_Float16*)d_ws;
    _Float16* wp = wq + 768 * 256;

    qwa_prep_weights<<<768, 256, 0, stream>>>(qkv_w, proj_w, wq, wp);
    qwa_fused<<<2048, 256, 0, stream>>>(x, mask, wq, qkv_b, wp, proj_b, rel_tab,
                                        s_q, s_k, s_v, s_attn, s_after,
                                        (float*)d_out);
}
